// VoxelBackBone8x_36438502539545
// MI455X (gfx1250) — matrix operand order repaired and checked by reading, not compiled, not run on hardware
//
#include <hip/hip_runtime.h>

typedef __attribute__((ext_vector_type(16))) _Float16 v16h;
typedef __attribute__((ext_vector_type(8)))  float    v8f;

#define MINVAR 2e-7f
#define BN_EPS 1e-3f

// ---------------------------------------------------------------------------
// Input init: mask = (mask_int==0), m = feat*mask, v = MINVAR*mask
// ---------------------------------------------------------------------------
__global__ void k_init(const float* __restrict__ feat, const int* __restrict__ mi,
                       float* __restrict__ m, float* __restrict__ v,
                       float* __restrict__ mask, int nvox) {
  for (int i = blockIdx.x * blockDim.x + threadIdx.x; i < nvox;
       i += gridDim.x * blockDim.x) {
    float nm = (mi[i] == 0) ? 1.f : 0.f;
    mask[i] = nm;
    #pragma unroll
    for (int c = 0; c < 4; ++c) {
      m[i * 4 + c] = feat[i * 4 + c] * nm;
      v[i * 4 + c] = MINVAR * nm;
    }
  }
}

// ---------------------------------------------------------------------------
// Weight prep: f32 W[K,Cout] -> transposed, zero-padded f16 W^T and (W^2)^T
// layout [Cout][Kpad] so B fragments are two contiguous 8-half runs per lane.
// ---------------------------------------------------------------------------
__global__ void k_wprep_t(const float* __restrict__ W, _Float16* __restrict__ WhT,
                          _Float16* __restrict__ W2hT, int Ktot, int Cout, int Kpad) {
  int total = Cout * Kpad;
  for (int i = blockIdx.x * blockDim.x + threadIdx.x; i < total;
       i += gridDim.x * blockDim.x) {
    int n = i / Kpad;
    int k = i - n * Kpad;
    float w = (k < Ktot) ? W[k * Cout + n] : 0.f;
    WhT[i]  = (_Float16)w;
    W2hT[i] = (_Float16)(w * w);
  }
}

// ---------------------------------------------------------------------------
// Mask dilation for strided (non-submanifold) layers
// ---------------------------------------------------------------------------
__global__ void k_maskconv(const float* __restrict__ min_, float* __restrict__ mout,
                           int inD, int inH, int inW, int outD, int outH, int outW,
                           int kd, int kh, int kw, int sz, int sy, int sx,
                           int pz, int py, int px) {
  int N = outD * outH * outW;
  for (int o = blockIdx.x * blockDim.x + threadIdx.x; o < N;
       o += gridDim.x * blockDim.x) {
    int oz = o / (outH * outW);
    int r  = o - oz * outH * outW;
    int oy = r / outW, ox = r - (r / outW) * outW;
    float acc = 0.f;
    for (int kz = 0; kz < kd; ++kz) {
      int iz = oz * sz - pz + kz; if (iz < 0 || iz >= inD) continue;
      for (int ky = 0; ky < kh; ++ky) {
        int iy = oy * sy - py + ky; if (iy < 0 || iy >= inH) continue;
        for (int kx = 0; kx < kw; ++kx) {
          int ix = ox * sx - px + kx; if (ix < 0 || ix >= inW) continue;
          acc += min_[((long)iz * inH + iy) * inW + ix];
        }
      }
    }
    mout[o] = (acc > 0.f) ? 1.f : 0.f;
  }
}

// ---------------------------------------------------------------------------
// Implicit-GEMM ADF conv via WMMA f32_16x16x32_f16.
//   Block = 128 threads (4 waves), tile = 32 voxels x 32 cout.
//   K-loop steps by 64 (two WMMA sub-chunks): 4 WMMAs/iter per wave.
//   A (im2col of m and v) staged in LDS as f16, b128-vectorized.
//   B read straight from pre-transposed, K-padded global f16 weights.
// ---------------------------------------------------------------------------
__global__ void __launch_bounds__(128)
k_conv_wmma(const float* __restrict__ min_, const float* __restrict__ vin_,
            const _Float16* __restrict__ WhT, const _Float16* __restrict__ W2hT,
            float* __restrict__ mout, float* __restrict__ vout,
            int inD, int inH, int inW, int outD, int outH, int outW,
            int Cout, int lcin, int cmask, int kd, int kh, int kw,
            int sz, int sy, int sx, int pz, int py, int px, int Ktot, int Kpad) {
  const int NvoxOut = outD * outH * outW;
  const int nTilesN = (Cout + 31) / 32;
  const int tileN   = blockIdx.x % nTilesN;
  const int tileM   = blockIdx.x / nTilesN;
  const int tid     = threadIdx.x;
  const int lane    = tid & 31;
  const int wave    = tid >> 5;
  const int wm      = wave >> 1;   // 16-voxel half of the 32-voxel tile
  const int wn      = wave & 1;    // 16-channel half of the 32-channel tile

  __shared__ _Float16 sAm[32][72];  // [local voxel][K 0..63], +8 pad
  __shared__ _Float16 sAv[32][72];

  // Staging map: each thread fills 16 contiguous cols of one row (b128 x2).
  const int lrow  = tid >> 2;        // 0..31
  const int jbase = (tid & 3) * 16;  // 0,16,32,48

  const int ov = tileM * 32 + lrow;
  const bool ovok = (ov < NvoxOut);
  int oz = 0, oy = 0, ox = 0;
  if (ovok) {
    oz = ov / (outH * outW);
    int r = ov - oz * outH * outW;
    oy = r / outW;
    ox = r - oy * outW;
  }
  const int bz = oz * sz - pz, by = oy * sy - py, bx = ox * sx - px;
  const int khw = kh * kw;

  const int g  = lane >> 4, ml = lane & 15;
  const int nglob = tileN * 32 + wn * 16 + ml;
  const bool nok  = (nglob < Cout);  // wave-uniform (Cout multiple of 16)
  const _Float16* bptr  = WhT  + (long)nglob * Kpad;
  const _Float16* b2ptr = W2hT + (long)nglob * Kpad;

  v8f accm = {}, accv = {};
  union V16 { v16h v; _Float16 e[16]; };
  union V8  { v8f  v; float    e[8];  };

  for (int k0 = 0; k0 < Kpad; k0 += 64) {
    // ---- stage A tiles (im2col gather, f32 -> f16) ----
    _Float16 tm[16], tv[16];
    #pragma unroll
    for (int e = 0; e < 16; ++e) {
      int k = k0 + jbase + e;
      float am = 0.f, av = 0.f;
      if (ovok && k < Ktot) {
        int tap = k >> lcin;
        int cin = k & cmask;
        int kz  = tap / khw;
        int r2  = tap - kz * khw;
        int ky  = r2 / kw;
        int kx  = r2 - ky * kw;
        int iz = bz + kz, iy = by + ky, ix = bx + kx;
        if ((unsigned)iz < (unsigned)inD && (unsigned)iy < (unsigned)inH &&
            (unsigned)ix < (unsigned)inW) {
          long idx = ((((long)iz * inH + iy) * inW + ix) << lcin) + cin;
          am = min_[idx];
          av = vin_[idx];
        }
      }
      tm[e] = (_Float16)am;
      tv[e] = (_Float16)av;
    }
    #pragma unroll
    for (int e = 0; e < 16; ++e) {
      sAm[lrow][jbase + e] = tm[e];
      sAv[lrow][jbase + e] = tv[e];
    }
    __syncthreads();

    // ---- B fragments straight from global (contiguous per lane) ----
    V16 fb0, fb1, fb20, fb21;
    #pragma unroll
    for (int i = 0; i < 16; ++i) {
      int kr = 8 * g + i + ((i >= 8) ? 8 : 0);  // lanes 0-15: K=0..7,16..23; lanes 16-31: K=8..15,24..31 of sub-chunk
      fb0.e[i]  = nok ? bptr [k0 + kr]      : (_Float16)0.f;
      fb1.e[i]  = nok ? bptr [k0 + 32 + kr] : (_Float16)0.f;
      fb20.e[i] = nok ? b2ptr[k0 + kr]      : (_Float16)0.f;
      fb21.e[i] = nok ? b2ptr[k0 + 32 + kr] : (_Float16)0.f;
    }

    // ---- A fragments from LDS (two contiguous 8-half runs each) ----
    V16 fa0, fa1, fv0, fv1;
    const int arow = wm * 16 + ml;
    #pragma unroll
    for (int i = 0; i < 16; ++i) {
      int kc = 8 * g + i + ((i >= 8) ? 8 : 0);
      fa0.e[i] = sAm[arow][kc];
      fa1.e[i] = sAm[arow][32 + kc];
      fv0.e[i] = sAv[arow][kc];
      fv1.e[i] = sAv[arow][32 + kc];
    }

    accm = __builtin_amdgcn_wmma_f32_16x16x32_f16(
        false, fa0.v, false, fb0.v,  (short)0, accm, false, false);
    accm = __builtin_amdgcn_wmma_f32_16x16x32_f16(
        false, fa1.v, false, fb1.v,  (short)0, accm, false, false);
    accv = __builtin_amdgcn_wmma_f32_16x16x32_f16(
        false, fv0.v, false, fb20.v, (short)0, accv, false, false);
    accv = __builtin_amdgcn_wmma_f32_16x16x32_f16(
        false, fv1.v, false, fb21.v, (short)0, accv, false, false);
    __syncthreads();
  }

  // ---- store C tiles (lane l: N = l%16; VGPR r: M = r + 8*(l/16)) ----
  V8 um, uv;
  um.v = accm; uv.v = accv;
  if (nok) {
    #pragma unroll
    for (int r = 0; r < 8; ++r) {
      int o = tileM * 32 + wm * 16 + 8 * g + r;
      if (o < NvoxOut) {
        long oi = (long)o * Cout + nglob;
        mout[oi] = um.e[r];
        vout[oi] = uv.e[r];
      }
    }
  }
}

// ---------------------------------------------------------------------------
// BN stats pass 1: 64 blocks, channel-interleaved (coalesced), deterministic.
//   thread c = tid & (Cout-1); partials per block.
// ---------------------------------------------------------------------------
__global__ void __launch_bounds__(256)
k_stats1(const float* __restrict__ m, const float* __restrict__ mask,
         float* __restrict__ p1, float* __restrict__ p2, float* __restrict__ pc,
         int Nvox, int Cout, int lcout) {
  int tid = threadIdx.x;
  int c   = tid & (Cout - 1);
  int grp = tid >> lcout;
  int gpb = 256 >> lcout;                 // groups per block
  int gid = blockIdx.x * gpb + grp;
  int stride = gridDim.x * gpb;
  float s1 = 0.f, s2 = 0.f, cnt = 0.f;
  for (int v = gid; v < Nvox; v += stride) {
    float nm = mask[v];
    float x  = m[(long)v * Cout + c];
    s1 += x * nm;
    s2 += x * x * nm;
    if (c == 0) cnt += nm;
  }
  __shared__ float r1[256], r2[256], rc[256];
  r1[tid] = s1; r2[tid] = s2; rc[tid] = (c == 0) ? cnt : 0.f;
  __syncthreads();
  for (int off = 128; off >= Cout; off >>= 1) {
    if (tid < off) { r1[tid] += r1[tid + off]; r2[tid] += r2[tid + off]; rc[tid] += rc[tid + off]; }
    __syncthreads();
  }
  if (tid < Cout) {
    p1[blockIdx.x * 128 + tid] = r1[tid];
    p2[blockIdx.x * 128 + tid] = r2[tid];
  }
  if (tid == 0) pc[blockIdx.x] = rc[0];
}

// ---------------------------------------------------------------------------
// BN stats pass 2: combine 64 block partials (single block, deterministic).
//   stats[0]=cnt, stats[1+c]=sum(m*nm), stats[129+c]=sum(m^2*nm)
// ---------------------------------------------------------------------------
__global__ void __launch_bounds__(256)
k_stats2(const float* __restrict__ p1, const float* __restrict__ p2,
         const float* __restrict__ pc, float* __restrict__ stats, int Cout, int nb) {
  int tid = threadIdx.x;
  if (tid < Cout) {
    float s1 = 0.f, s2 = 0.f;
    for (int b = 0; b < nb; ++b) { s1 += p1[b * 128 + tid]; s2 += p2[b * 128 + tid]; }
    stats[1 + tid]   = s1;
    stats[129 + tid] = s2;
  }
  if (tid == 0) {
    float cnt = 0.f;
    for (int b = 0; b < nb; ++b) cnt += pc[b];
    stats[0] = cnt;
  }
}

// ---------------------------------------------------------------------------
// Fused BN normalize + ADF Gaussian ReLU + mask, in place.
// ---------------------------------------------------------------------------
__global__ void k_point(float* __restrict__ m, float* __restrict__ v,
                        const float* __restrict__ mask, const float* __restrict__ stats,
                        const float* __restrict__ gamma, const float* __restrict__ beta,
                        int Nvox, int Cout) {
  long total = (long)Nvox * Cout;
  float cnt = fmaxf(stats[0], 1.f);
  for (long i = (long)blockIdx.x * blockDim.x + threadIdx.x; i < total;
       i += (long)gridDim.x * blockDim.x) {
    int  c  = (int)(i % Cout);
    long vx = i / Cout;
    float mu  = stats[1 + c] / cnt;
    float var = stats[129 + c] / cnt - mu * mu;
    float inv = gamma[c] * rsqrtf(var + BN_EPS);
    float mn  = (m[i] - mu) * inv + beta[c];
    float vn  = fmaxf(v[i] * inv * inv, 0.f) + MINVAR;
    float sg  = sqrtf(vn);
    float r   = mn / sg;
    float cdf = 0.5f * (1.f + erff(r * 0.70710678118654752f));
    float pdf = 0.39894228040143270f * __expf(-0.5f * r * r);
    float mo  = mn * cdf + sg * pdf;
    float vo  = fmaxf((mn * mn + vn) * cdf + mn * sg * pdf - mo * mo, 0.f) + MINVAR;
    float nm  = mask[vx];
    m[i] = mo * nm;
    v[i] = vo * nm;
  }
}

// ---------------------------------------------------------------------------
// Final concat: out[vox, 0:128]=m, out[vox, 128:256]=v
// ---------------------------------------------------------------------------
__global__ void k_concat(const float* __restrict__ m, const float* __restrict__ v,
                         float* __restrict__ out, int Nvox, int C) {
  int total = Nvox * 2 * C;
  for (int i = blockIdx.x * blockDim.x + threadIdx.x; i < total;
       i += gridDim.x * blockDim.x) {
    int c  = i % (2 * C);
    int vx = i / (2 * C);
    out[i] = (c < C) ? m[(long)vx * C + c] : v[(long)vx * C + c - C];
  }
}

// ---------------------------------------------------------------------------
// Host orchestration
// ---------------------------------------------------------------------------
struct LD { int cin, cout, kd, kh, kw, sz, sy, sx, pz, py, px, subm; };
static const LD g_layers[12] = {
  {  4,  16, 3,3,3, 1,1,1, 1,1,1, 1},
  { 16,  16, 3,3,3, 1,1,1, 1,1,1, 1},
  { 16,  32, 3,3,3, 2,2,2, 1,1,1, 0},
  { 32,  32, 3,3,3, 1,1,1, 1,1,1, 1},
  { 32,  32, 3,3,3, 1,1,1, 1,1,1, 1},
  { 32,  64, 3,3,3, 2,2,2, 1,1,1, 0},
  { 64,  64, 3,3,3, 1,1,1, 1,1,1, 1},
  { 64,  64, 3,3,3, 1,1,1, 1,1,1, 1},
  { 64,  64, 3,3,3, 2,2,2, 0,1,1, 0},
  { 64,  64, 3,3,3, 1,1,1, 1,1,1, 1},
  { 64,  64, 3,3,3, 1,1,1, 1,1,1, 1},
  { 64, 128, 3,1,1, 2,1,1, 0,0,0, 0},
};

static inline int ilog2(int x) { int l = 0; while ((1 << l) < x) ++l; return l; }

extern "C" void kernel_launch(void* const* d_in, const int* in_sizes, int n_in,
                              void* d_out, int out_size, void* d_ws, size_t ws_size,
                              hipStream_t stream) {
  (void)in_sizes; (void)n_in; (void)out_size; (void)ws_size;
  const float* feat = (const float*)d_in[0];
  const int*   mi   = (const int*)d_in[1];

  const size_t NV0  = (size_t)41 * 200 * 176;  // 1,443,200 voxels at full res
  const size_t MAXE = NV0 * 16;                // largest m/v tensor (layers 0/1)
  const int    NSB  = 64;                      // stats pass-1 blocks

  float* base  = (float*)d_ws;
  float* mA    = base;
  float* mB    = mA + MAXE;
  float* vA    = mB + MAXE;
  float* vB    = vA + MAXE;
  float* maskA = vB + MAXE;
  float* maskB = maskA + NV0;
  float* stats = maskB + NV0;                       // 257 floats used
  float* part1 = stats + 272;                       // 64 * 128
  float* part2 = part1 + NSB * 128;                 // 64 * 128
  float* partc = part2 + NSB * 128;                 // 64
  _Float16* WhT  = (_Float16*)(partc + 64);
  _Float16* W2hT = WhT + (size_t)128 * 1792;        // max Cout*Kpad = 64*1728, fits

  int D = 41, H = 200, W = 176;
  k_init<<<(int)((NV0 + 255) / 256), 256, 0, stream>>>(feat, mi, mA, vA, maskA, (int)NV0);

  float *mcur = mA, *vcur = vA, *mnxt = mB, *vnxt = vB;
  float *mk = maskA, *mko = maskB;

  for (int i = 0; i < 12; ++i) {
    const LD& L = g_layers[i];
    int oD = (D + 2 * L.pz - L.kd) / L.sz + 1;
    int oH = (H + 2 * L.py - L.kh) / L.sy + 1;
    int oW = (W + 2 * L.px - L.kw) / L.sx + 1;
    int Nout = oD * oH * oW;
    int Ktot = L.kd * L.kh * L.kw * L.cin;
    int Kpad = ((Ktot + 63) / 64) * 64;
    int lcin = ilog2(L.cin);
    int lcout = ilog2(L.cout);

    const float* Wp = (const float*)d_in[2 + 3 * i];
    const float* gp = (const float*)d_in[3 + 3 * i];
    const float* bp = (const float*)d_in[4 + 3 * i];

    k_wprep_t<<<(L.cout * Kpad + 255) / 256, 256, 0, stream>>>(
        Wp, WhT, W2hT, Ktot, L.cout, Kpad);

    float* nmask = mk;
    if (!L.subm) {
      k_maskconv<<<(Nout + 255) / 256, 256, 0, stream>>>(
          mk, mko, D, H, W, oD, oH, oW,
          L.kd, L.kh, L.kw, L.sz, L.sy, L.sx, L.pz, L.py, L.px);
      nmask = mko;
    }

    int nTN    = (L.cout + 31) / 32;
    int tilesM = (Nout + 31) / 32;
    k_conv_wmma<<<dim3(tilesM * nTN), 128, 0, stream>>>(
        mcur, vcur, WhT, W2hT, mnxt, vnxt,
        D, H, W, oD, oH, oW, L.cout, lcin, L.cin - 1,
        L.kd, L.kh, L.kw, L.sz, L.sy, L.sx, L.pz, L.py, L.px, Ktot, Kpad);

    k_stats1<<<NSB, 256, 0, stream>>>(mnxt, nmask, part1, part2, partc,
                                      Nout, L.cout, lcout);
    k_stats2<<<1, 256, 0, stream>>>(part1, part2, partc, stats, L.cout, NSB);

    long tot = (long)Nout * L.cout;
    int pgrid = (int)((tot + 255) / 256);
    if (pgrid > 65535) pgrid = 65535;
    k_point<<<pgrid, 256, 0, stream>>>(mnxt, vnxt, nmask, stats, gp, bp, Nout, L.cout);

    float* t;
    t = mcur; mcur = mnxt; mnxt = t;
    t = vcur; vcur = vnxt; vnxt = t;
    if (!L.subm) { t = mk; mk = mko; mko = t; }
    D = oD; H = oH; W = oW;
  }

  // Final: [2,25,22] voxels, 128 channels of m then 128 of v.
  int NvoxF = D * H * W;  // 1100
  k_concat<<<(NvoxF * 256 + 255) / 256, 256, 0, stream>>>(mcur, vcur, (float*)d_out, NvoxF, 128);
}
